// VictorDynamicNeuromorphicAttention_30365418782774
// MI455X (gfx1250) — compile-verified, matrix operand order checked by reading, not run on hardware
//
#include <hip/hip_runtime.h>
#include <hip/hip_bf16.h>

typedef _Float16 v16h __attribute__((ext_vector_type(16)));
typedef _Float16 v8h  __attribute__((ext_vector_type(8)));
typedef _Float16 v4h  __attribute__((ext_vector_type(4)));
typedef float    v8f  __attribute__((ext_vector_type(8)));
typedef unsigned int u32x4 __attribute__((ext_vector_type(4)));
typedef int          i32x8 __attribute__((ext_vector_type(8)));
typedef int          i32x4 __attribute__((ext_vector_type(4)));

#if __has_builtin(__builtin_amdgcn_tensor_load_to_lds)
#define USE_TDM 1
#else
#define USE_TDM 0
#endif

// LDS B tile: 64 rows x 128 halves per chunk, padded to 136 halves (272B) per
// row so consecutive fragment rows hit different LDS banks (256B stride would
// alias all 16 rows onto the same bank: 64 banks x 4B).
#define BROW 136
#define CHUNK 128

// Load one 16-half WMMA fragment (global, row stride = ld).
// 16-bit A/B^T layout: lane<16 holds K = k+0..7 and k+16..23,
// lane>=16 holds K = k+8..15 and k+24..31 (koff pre-applied by caller).
__device__ __forceinline__ v16h load_frag(const _Float16* __restrict__ p) {
  v8h lo = *(const v8h*)(p);
  v8h hi = *(const v8h*)(p + 16);
  return __builtin_shufflevector(lo, hi, 0,1,2,3,4,5,6,7,8,9,10,11,12,13,14,15);
}

__device__ __forceinline__ v16h lds_frag(const _Float16* p) {
  v8h lo = *(const v8h*)(p);
  v8h hi = *(const v8h*)(p + 16);
  return __builtin_shufflevector(lo, hi, 0,1,2,3,4,5,6,7,8,9,10,11,12,13,14,15);
}

#if USE_TDM
// Issue a TDM DMA of a 64-row x 128-half tile of Bt (row stride ldb halves)
// into LDS at lds_addr, with 4-DWORD padding every 64 DWORDs (-> 272B rows).
// This toolchain exposes the 6-arg clang-23 builtin:
//   (u32x4 group0, i32x8 group1, i32x4 group2, i32x4 group3, i32x8, i32 cpol)
__device__ __forceinline__ void tdm_load_B(const _Float16* gptr, unsigned lds_addr, int ldb) {
  unsigned long long ga = (unsigned long long)(uintptr_t)gptr;
  u32x4 g0;
  g0[0] = 1u;                                   // count=1 (valid user descriptor)
  g0[1] = lds_addr;                             // lds_addr [63:32]
  g0[2] = (unsigned)ga;                         // global_addr [95:64]
  g0[3] = (unsigned)(ga >> 32) | (2u << 30);    // global_addr hi + type=2 (image)
  i32x8 g1;
  // data_size=2B, pad_enable, pad_interval=64 DWORDs (code 5), pad_amount=4 DWORDs (code 3)
  g1[0] = (int)((1u << 16) | (1u << 20) | (5u << 22) | (3u << 25));
  const unsigned td0 = 0x40000000u, td1 = 0x40000000u;  // huge dims: no OOB clamp
  g1[1] = (int)((td0 & 0xFFFFu) << 16);                 // tensor_dim0[15:0]
  g1[2] = (int)((td0 >> 16) | ((td1 & 0xFFFFu) << 16)); // tensor_dim0 hi | tensor_dim1 lo
  g1[3] = (int)((td1 >> 16) | ((unsigned)CHUNK << 16)); // tensor_dim1 hi | tile_dim0=128
  g1[4] = 64;                                           // tile_dim1=64, tile_dim2=0
  g1[5] = ldb;                                          // tensor_dim0_stride[31:0]
  g1[6] = 0;                                            // stride hi | dim1_stride lo
  g1[7] = 0;
  i32x4 z4 = {0, 0, 0, 0};                              // group2/3: 2D tile -> zero (NULL)
  i32x8 z8 = {0, 0, 0, 0, 0, 0, 0, 0};
  __builtin_amdgcn_tensor_load_to_lds(g0, g1, z4, z4, z8, 0);
}
#endif

// Generic C = scale * (A[M,K] x Bt[N,K]^T + bias[col]) GEMM, f16 in, f32 acc.
// flags bit0: store f16 (else f32); bit1: store transposed (C[col*ldc+row]).
// Grid: x = N/64, y = M/256, z = batch. Block: 256 threads = 8 waves.
// The 64-col B panel is staged into LDS (TDM DMA, double buffered, 128-wide K
// chunks) and shared by all 8 waves; each wave computes a 32x64 sub-tile
// (2 A frags x 4 B frags -> 8 WMMAs per k-step).
__global__ void __launch_bounds__(256)
gemm_f16_wmma(const _Float16* __restrict__ A, long sAz, int lda,
              const _Float16* __restrict__ Bt, long sBz, int ldb,
              void* __restrict__ Cv, long sCz, int ldc,
              const float* __restrict__ bias, long sBiasz,
              float scale, int M, int N, int K, int flags)
{
  (void)M; (void)N;
  const int z    = blockIdx.z;
  A  += z * sAz;
  Bt += z * sBz;
  const int lane = threadIdx.x & 31;
  const int wave = threadIdx.x >> 5;
  const int m0   = blockIdx.y * 256 + wave * 32;
  const int n0   = blockIdx.x * 64;
  const int lr   = lane & 15;
  const int koff = (lane >> 4) * 8;

  __shared__ __align__(16) _Float16 tileB[2][64 * BROW];

  const _Float16* abase0 = A + (long)(m0 +  0 + lr) * lda + koff;
  const _Float16* abase1 = A + (long)(m0 + 16 + lr) * lda + koff;
  const _Float16* btile  = Bt + (long)n0 * ldb;   // 64-row panel for this block

  const int nchunks = K >> 7;   // K / 128

  // Prologue: stage chunk 0 into buffer 0.
#if USE_TDM
  if (threadIdx.x < 32) {
    tdm_load_B(btile, (unsigned)(uintptr_t)&tileB[0][0], ldb);
    __builtin_amdgcn_s_wait_tensorcnt(0);
  }
#else
  {
    const int tid = threadIdx.x;
#pragma unroll
    for (int t = 0; t < 4; ++t) {
      int idx = t * 256 + tid;          // 1024 16-byte chunks
      int r = idx >> 3, c8 = (idx & 7) * 16;
      *(v8h*)(&tileB[0][r * BROW + c8]) = *(const v8h*)(btile + (long)r * ldb + c8);
    }
  }
#endif
  __syncthreads();

  v8f acc[4][2] = {};
  for (int c = 0; c < nchunks; ++c) {
    const int buf = c & 1;
    const int kc  = c * CHUNK;
    const bool more = (c + 1) < nchunks;

    // Overlap: DMA chunk c+1 into the other buffer while computing chunk c.
#if USE_TDM
    if (more && threadIdx.x < 32)
      tdm_load_B(btile + kc + CHUNK, (unsigned)(uintptr_t)&tileB[buf ^ 1][0], ldb);
#else
    if (more) {
      const int tid = threadIdx.x;
#pragma unroll
      for (int t = 0; t < 4; ++t) {
        int idx = t * 256 + tid;
        int r = idx >> 3, c8 = (idx & 7) * 16;
        *(v8h*)(&tileB[buf ^ 1][r * BROW + c8]) =
            *(const v8h*)(btile + (long)r * ldb + kc + CHUNK + c8);
      }
    }
#endif
    // Keep A one chunk ahead in the WGP cache (speculative; OOB dropped).
    __builtin_prefetch((const void*)(abase0 + kc + CHUNK), 0, 1);
    __builtin_prefetch((const void*)(abase1 + kc + CHUNK), 0, 1);

    const _Float16* Bl = &tileB[buf][0];
#pragma unroll
    for (int kk = 0; kk < CHUNK; kk += 32) {
      v16h a0 = load_frag(abase0 + kc + kk);
      v16h a1 = load_frag(abase1 + kc + kk);
      v16h f0 = lds_frag(Bl + ( 0 + lr) * BROW + kk + koff);
      v16h f1 = lds_frag(Bl + (16 + lr) * BROW + kk + koff);
      v16h f2 = lds_frag(Bl + (32 + lr) * BROW + kk + koff);
      v16h f3 = lds_frag(Bl + (48 + lr) * BROW + kk + koff);
      acc[0][0] = __builtin_amdgcn_wmma_f32_16x16x32_f16(false, a0, false, f0, (short)0, acc[0][0], false, false);
      acc[0][1] = __builtin_amdgcn_wmma_f32_16x16x32_f16(false, a1, false, f0, (short)0, acc[0][1], false, false);
      acc[1][0] = __builtin_amdgcn_wmma_f32_16x16x32_f16(false, a0, false, f1, (short)0, acc[1][0], false, false);
      acc[1][1] = __builtin_amdgcn_wmma_f32_16x16x32_f16(false, a1, false, f1, (short)0, acc[1][1], false, false);
      acc[2][0] = __builtin_amdgcn_wmma_f32_16x16x32_f16(false, a0, false, f2, (short)0, acc[2][0], false, false);
      acc[2][1] = __builtin_amdgcn_wmma_f32_16x16x32_f16(false, a1, false, f2, (short)0, acc[2][1], false, false);
      acc[3][0] = __builtin_amdgcn_wmma_f32_16x16x32_f16(false, a0, false, f3, (short)0, acc[3][0], false, false);
      acc[3][1] = __builtin_amdgcn_wmma_f32_16x16x32_f16(false, a1, false, f3, (short)0, acc[3][1], false, false);
    }

    if (more) {
#if USE_TDM
      if (threadIdx.x < 32) __builtin_amdgcn_s_wait_tensorcnt(0);
#endif
      __syncthreads();   // publish buf^1; all waves done reading buf
    }
  }

  // Epilogue. C layout per 16x16 tile: lanes 0-15 -> M = base+0..7,
  // lanes 16-31 -> M = base+8..15, N = n0 + j*16 + (lane&15);
  // accumulator element r holds row (base + r).
  float*    Cf = (float*)Cv    + (long)z * sCz;
  _Float16* Ch = (_Float16*)Cv + (long)z * sCz;
  const float* bp = bias ? bias + (long)z * sBiasz : nullptr;
#pragma unroll
  for (int j = 0; j < 4; ++j) {
    const int col = n0 + j * 16 + lr;
    const float bval = bp ? bp[col] : 0.0f;
#pragma unroll
    for (int t = 0; t < 2; ++t) {
      const int rowb = m0 + t * 16 + (lane >> 4) * 8;
#pragma unroll
      for (int r = 0; r < 8; ++r) {
        float o = (acc[j][t][r] + bval) * scale;
        long addr = (flags & 2) ? ((long)col * ldc + (rowb + r))
                                : ((long)(rowb + r) * ldc + col);
        if (flags & 1) Ch[addr] = (_Float16)o;
        else           Cf[addr] = o;
      }
    }
  }
}

__global__ void cvt_f32_f16(const float* __restrict__ src, _Float16* __restrict__ dst, int n) {
  int i = blockIdx.x * blockDim.x + threadIdx.x;
  if (i < n) dst[i] = (_Float16)src[i];
}

// dst[c*ldDst + r] = src[r*C + c]  (f32 -> f16)
__global__ void transpose_f32_f16(const float* __restrict__ src, _Float16* __restrict__ dst,
                                  int R, int C, int ldDst) {
  int c = blockIdx.x * 16 + threadIdx.x;
  int r = blockIdx.y * 16 + threadIdx.y;
  if (r < R && c < C) dst[(long)c * ldDst + r] = (_Float16)src[(long)r * C + c];
}

__global__ void bo_sum_kernel(const float* __restrict__ bo, float* __restrict__ out) {
  int d = blockIdx.x * blockDim.x + threadIdx.x;
  if (d < 512) {
    float s = 0.0f;
    for (int n = 0; n < 4; ++n) s += bo[n * 512 + d];
    out[d] = s;
  }
}

// One block per score row: P_row = sum_h gate_h * softmax(s_h * row).
// Row max commutes with the positive per-head scale, so one max serves all heads.
__global__ void __launch_bounds__(256)
softmax_combine(const float* __restrict__ scores, _Float16* __restrict__ P,
                const float* __restrict__ strength, const float* __restrict__ gate)
{
  const int S = 1024;
  const int row = blockIdx.x;
  const int zb  = blockIdx.y;     // n*2 + b
  const int n   = zb >> 1;
  const float4* src = (const float4*)(scores + ((long)zb * S + row) * S);
  _Float16*     dst = P + ((long)zb * S + row) * S;
  const int tid = threadIdx.x;    // 0..255, 4 elements each

  __shared__ float red[256];
  __shared__ float wsh[8];

  float4 xv = src[tid];
  float m = fmaxf(fmaxf(xv.x, xv.y), fmaxf(xv.z, xv.w));
  red[tid] = m; __syncthreads();
  for (int s = 128; s > 0; s >>= 1) {
    if (tid < s) red[tid] = fmaxf(red[tid], red[tid + s]);
    __syncthreads();
  }
  m = red[0];

  float sh[8], gh[8];
#pragma unroll
  for (int h = 0; h < 8; ++h) {
    sh[h] = fminf(fmaxf(strength[n * 64 + h * 9], 0.01f), 1.0f);  // diag, clamped
    gh[h] = gate[n * 8 + h];
  }

  for (int h = 0; h < 8; ++h) {
    float s = sh[h];
    float t = __expf(s * (xv.x - m)) + __expf(s * (xv.y - m)) +
              __expf(s * (xv.z - m)) + __expf(s * (xv.w - m));
    __syncthreads();
    red[tid] = t; __syncthreads();
    for (int st = 128; st > 0; st >>= 1) {
      if (tid < st) red[tid] += red[tid + st];
      __syncthreads();
    }
    if (tid == 0) wsh[h] = gh[h] / red[0];
  }
  __syncthreads();

  float o0 = 0, o1 = 0, o2 = 0, o3 = 0;
#pragma unroll
  for (int h = 0; h < 8; ++h) {
    float s = sh[h], wv = wsh[h];
    o0 += wv * __expf(s * (xv.x - m));
    o1 += wv * __expf(s * (xv.y - m));
    o2 += wv * __expf(s * (xv.z - m));
    o3 += wv * __expf(s * (xv.w - m));
  }
  v4h o = {(_Float16)o0, (_Float16)o1, (_Float16)o2, (_Float16)o3};
  *(v4h*)(dst + tid * 4) = o;
}

extern "C" void kernel_launch(void* const* d_in, const int* in_sizes, int n_in,
                              void* d_out, int out_size, void* d_ws, size_t ws_size,
                              hipStream_t stream) {
  (void)in_sizes; (void)n_in; (void)out_size; (void)ws_size;
  const float* x        = (const float*)d_in[0];
  const float* Wq       = (const float*)d_in[1];
  const float* bq       = (const float*)d_in[2];
  const float* Wk       = (const float*)d_in[3];
  const float* bk       = (const float*)d_in[4];
  const float* Wv       = (const float*)d_in[5];
  const float* bv       = (const float*)d_in[6];
  const float* Wo       = (const float*)d_in[7];
  const float* bo       = (const float*)d_in[8];
  const float* strength = (const float*)d_in[9];
  const float* gate     = (const float*)d_in[10];
  float* out = (float*)d_out;

  const int NB = 4, B = 2, S = 1024, D = 512, HID = 512;
  const long MS = (long)B * S;  // 2048

  char* w = (char*)d_ws;
  _Float16* x16  = (_Float16*)w; w += MS * D * 2;                  // [2048,512]
  _Float16* WqT  = (_Float16*)w; w += (long)NB * HID * D * 2;      // [n][HID,D]
  _Float16* WkT  = (_Float16*)w; w += (long)NB * HID * D * 2;
  _Float16* WvT  = (_Float16*)w; w += (long)NB * HID * D * 2;
  _Float16* WoT  = (_Float16*)w; w += (long)D * NB * HID * 2;      // [D, NB*HID]
  _Float16* q16  = (_Float16*)w; w += (long)NB * MS * HID * 2;     // [n][2048,512]
  _Float16* k16  = (_Float16*)w; w += (long)NB * MS * HID * 2;
  _Float16* vT16 = (_Float16*)w; w += (long)NB * B * HID * S * 2;  // [n][b][HID,S]
  float*    sc   = (float*)w;    w += (long)NB * B * S * S * 4;    // [n][b][S,S]
  _Float16* P16  = (_Float16*)w; w += (long)NB * B * S * S * 2;    // [n][b][S,S]
  _Float16* hs   = (_Float16*)w; w += MS * (long)NB * HID * 2;     // [b*S, NB*HID]
  float*    bos  = (float*)w;    w += 512 * 4;

  const dim3 tb(256);

  // Stage 0: f16 conversion + weight transposes (A x B^T convention everywhere).
  cvt_f32_f16<<<dim3((MS * D) / 256), 256, 0, stream>>>(x, x16, (int)(MS * D));
  for (int n = 0; n < NB; ++n) {
    transpose_f32_f16<<<dim3(32, 32), dim3(16, 16), 0, stream>>>(
        Wq + (long)n * D * HID, WqT + (long)n * HID * D, D, HID, D);
    transpose_f32_f16<<<dim3(32, 32), dim3(16, 16), 0, stream>>>(
        Wk + (long)n * D * HID, WkT + (long)n * HID * D, D, HID, D);
    transpose_f32_f16<<<dim3(32, 32), dim3(16, 16), 0, stream>>>(
        Wv + (long)n * D * HID, WvT + (long)n * HID * D, D, HID, D);
    // WoT[d][n*HID + h] = Wo[n][h][d]
    transpose_f32_f16<<<dim3(32, 32), dim3(16, 16), 0, stream>>>(
        Wo + (long)n * HID * D, WoT + (long)n * HID, HID, D, NB * HID);
  }
  bo_sum_kernel<<<2, 256, 0, stream>>>(bo, bos);

  // Stage 1: Q, K projections: [2048,512] x [512,512]^T per block.
  gemm_f16_wmma<<<dim3(HID / 64, MS / 256, NB), tb, 0, stream>>>(
      x16, 0L, D, WqT, (long)HID * D, D,
      q16, (long)MS * HID, HID, bq, (long)HID, 1.0f, (int)MS, HID, D, 1);
  gemm_f16_wmma<<<dim3(HID / 64, MS / 256, NB), tb, 0, stream>>>(
      x16, 0L, D, WkT, (long)HID * D, D,
      k16, (long)MS * HID, HID, bk, (long)HID, 1.0f, (int)MS, HID, D, 1);

  // Stage 2: V projection, epilogue-transposed to vT[n][b][h][t].
  for (int n = 0; n < NB; ++n) {
    gemm_f16_wmma<<<dim3(HID / 64, S / 256, B), tb, 0, stream>>>(
        x16, (long)S * D, D, WvT + (long)n * HID * D, 0L, D,
        vT16 + (long)n * B * HID * S, (long)HID * S, S,
        bv + (long)n * HID, 0L, 1.0f, S, HID, D, 3);
  }

  // Stage 3: scores = q k^T / sqrt(512); q/k offsets are linear in z = n*2+b.
  gemm_f16_wmma<<<dim3(S / 64, S / 256, NB * B), tb, 0, stream>>>(
      q16, (long)S * HID, HID, k16, (long)S * HID, HID,
      sc, (long)S * S, S, nullptr, 0L, 0.044194173824159216f, S, S, HID, 0);

  // Stage 4: combined gate-weighted softmax over all 8 heads -> one prob matrix.
  softmax_combine<<<dim3(S, NB * B), 256, 0, stream>>>(sc, P16, strength, gate);

  // Stage 5: head_sum = P @ v, staged as hs[b*S + s][n*HID + h].
  for (int n = 0; n < NB; ++n) {
    gemm_f16_wmma<<<dim3(HID / 64, S / 256, B), tb, 0, stream>>>(
        P16 + (long)n * B * S * S, (long)S * S, S,
        vT16 + (long)n * B * HID * S, (long)HID * S, S,
        hs + (long)n * HID, (long)S * NB * HID, NB * HID,
        nullptr, 0L, 1.0f, S, HID, S, 1);
  }

  // Stage 6: out = MOD * (hs @ Wo_all + sum_n bo[n]); K = NB*HID = 2048.
  gemm_f16_wmma<<<dim3(D / 64, MS / 256, 1), tb, 0, stream>>>(
      hs, 0L, NB * HID, WoT, 0L, NB * HID,
      out, 0L, D, bos, 0L, 1.55f, (int)MS, D, NB * HID, 0);
}
